// MixtureOfExperts_42167988912991
// MI455X (gfx1250) — compile-verified, matrix operand order checked by reading, not run on hardware
//
#include <hip/hip_runtime.h>
#include <hip/hip_bf16.h>
#include <math.h>

#define D_MODEL 512
#define D_FF    2048
#define N_EXP   8
#define N_TOK   8192
#define M_TILE  64
#define F_TILE  128
#define KSTEP   32

// LDS strides (bf16 elements). Row byte offsets stay 16B aligned and are not a
// multiple of the 256B bank period. These strides are reproduced by the TDM via
// pad_interval/pad_amount (see tdm_load_2d callers).
#define SX_STRIDE 520   // 64  x (512+8)
#define SH_STRIDE 136   // 64  x (128+8)
#define SW_STRIDE 40    // <=512 x (32+8)

typedef __attribute__((ext_vector_type(16))) __bf16 v16bf;
typedef __attribute__((ext_vector_type(8)))  __bf16 v8bf;
typedef __attribute__((ext_vector_type(8)))  float  v8f;
typedef __attribute__((ext_vector_type(4)))  unsigned int v4u;
typedef __attribute__((ext_vector_type(8)))  int v8i;
typedef __attribute__((ext_vector_type(4)))  int v4i;

__device__ __forceinline__ unsigned short f2bf(float f) {
  union { float f; unsigned int u; } c; c.f = f;
  unsigned int u = c.u;
  return (unsigned short)((u + 0x7FFFu + ((u >> 16) & 1u)) >> 16);  // RNE
}

// A/B fragment for v_wmma_f32_16x16x32_bf16, per ISA VGPR layout:
// VGPR0..3 = K kbase..kbase+7, VGPR4..7 = K 16+kbase.., kbase = (lane<16?0:8).
// p0 = &row_base[k0 + kbase]; rows and k0 are 16B aligned.
__device__ __forceinline__ v16bf load_frag(const unsigned short* p0) {
  v8bf lo = *(const v8bf*)p0;
  v8bf hi = *(const v8bf*)(p0 + 16);
  v16bf r;
#pragma unroll
  for (int i = 0; i < 8; ++i) { r[i] = lo[i]; r[i + 8] = hi[i]; }
  return r;
}

__device__ __forceinline__ v8f vzero8() {
  v8f z;
#pragma unroll
  for (int i = 0; i < 8; ++i) z[i] = 0.0f;
  return z;
}

// ---------------------------------------------------------------------------
// Tensor Data Mover: 2D strided tile (bf16) global -> LDS with row padding.
// D# built per CDNA5 ISA 8.3/8.4.  pad_icode: threshold = 2<<code DWORDs.
// pad_dw: padding amount in DWORDs (>=1).
// ---------------------------------------------------------------------------
__device__ __forceinline__ void tdm_load_2d(unsigned lds_off, const void* gptr,
                                            unsigned tensor_d0, unsigned tensor_d1,
                                            unsigned tile_d0, unsigned tile_d1,
                                            unsigned stride0,
                                            unsigned pad_icode, unsigned pad_dw) {
  unsigned long long ga = (unsigned long long)gptr;
  v4u g0;
  g0[0] = 1u;                                   // count=1, user descriptor
  g0[1] = lds_off;                              // LDS byte address
  g0[2] = (unsigned)ga;                         // global_addr[31:0]
  g0[3] = (unsigned)((ga >> 32) & 0x01FFFFFFu)  // global_addr[56:32]
          | (2u << 30);                         // type=2 ("image")
  v8i g1;
  g1[0] = (int)((1u << 16)                      // data_size = 2 bytes
                | (1u << 20)                    // pad_enable
                | (pad_icode << 22)             // pad_interval code
                | ((pad_dw - 1u) << 25));       // pad_amount (0 => 1 DWORD)
  g1[1] = (int)((tensor_d0 & 0xFFFFu) << 16);   // [63:48] tensor_dim0 lo16
  g1[2] = (int)((tensor_d0 >> 16) | ((tensor_d1 & 0xFFFFu) << 16));
  g1[3] = (int)((tensor_d1 >> 16) | (tile_d0 << 16));
  g1[4] = (int)tile_d1;                         // tile_dim1; tile_dim2 = 0
  g1[5] = (int)stride0;                         // tensor_dim0_stride lo32
  g1[6] = 0;
  g1[7] = 0;
  v4i gz; gz[0] = gz[1] = gz[2] = gz[3] = 0;
#if defined(__clang_major__) && (__clang_major__ >= 23)
  v8i gz8;
#pragma unroll
  for (int i = 0; i < 8; ++i) gz8[i] = 0;
  __builtin_amdgcn_tensor_load_to_lds(g0, g1, gz, gz, gz8, 0);
#else
  __builtin_amdgcn_tensor_load_to_lds(g0, g1, gz, gz, 0);
#endif
}

// ---------------------------------------------------------------------------
// Kernel 1: zero the final-output accumulation region.
// ---------------------------------------------------------------------------
__global__ void moe_zero_kernel(float4* __restrict__ out, int n4) {
  int i = blockIdx.x * blockDim.x + threadIdx.x;
  if (i < n4) out[i] = make_float4(0.f, 0.f, 0.f, 0.f);
}

// ---------------------------------------------------------------------------
// Kernel 2: router softmax (one thread per token).
// ---------------------------------------------------------------------------
__global__ void moe_router_kernel(const float* __restrict__ x,
                                  const float* __restrict__ wr,
                                  const float* __restrict__ br,
                                  float* __restrict__ rw) {
  int t = blockIdx.x * blockDim.x + threadIdx.x;
  if (t >= N_TOK) return;
  float acc[N_EXP];
#pragma unroll
  for (int e = 0; e < N_EXP; ++e) acc[e] = br[e];
  const float* xr = x + (size_t)t * D_MODEL;
  for (int d = 0; d < D_MODEL; ++d) {
    float xv = xr[d];
#pragma unroll
    for (int e = 0; e < N_EXP; ++e) acc[e] = fmaf(xv, wr[d * N_EXP + e], acc[e]);
  }
  float mx = acc[0];
#pragma unroll
  for (int e = 1; e < N_EXP; ++e) mx = fmaxf(mx, acc[e]);
  float s = 0.f;
#pragma unroll
  for (int e = 0; e < N_EXP; ++e) { acc[e] = expf(acc[e] - mx); s += acc[e]; }
  float inv = 1.0f / s;
#pragma unroll
  for (int e = 0; e < N_EXP; ++e) rw[t * N_EXP + e] = acc[e] * inv;
}

// ---------------------------------------------------------------------------
// Kernel 3a: x (f32) -> bf16 workspace (row-major [tok][d], pair-packed).
// ---------------------------------------------------------------------------
__global__ void moe_cvt_x_kernel(const float* __restrict__ x,
                                 unsigned int* __restrict__ xb2, int npair) {
  int i = blockIdx.x * blockDim.x + threadIdx.x;
  if (i >= npair) return;
  const float* s = x + (size_t)i * 2;
  xb2[i] = (unsigned int)f2bf(s[0]) | ((unsigned int)f2bf(s[1]) << 16);
}

// ---------------------------------------------------------------------------
// Kernel 3b: transpose+convert weights.
//   w1t[e][f][d] = bf16(w1[e][d][f])   (8 x 2048 x 512)
//   w2t[e][d][f] = bf16(w2[e][f][d])   (8 x 512 x 2048)
// One-time cost; strided reads, coalesced-ish writes.
// ---------------------------------------------------------------------------
__global__ void moe_cvt_w_kernel(const float* __restrict__ w1,
                                 const float* __restrict__ w2,
                                 unsigned short* __restrict__ w1t,
                                 unsigned short* __restrict__ w2t, int n) {
  int o = blockIdx.x * blockDim.x + threadIdx.x;
  if (o >= n) return;
  {
    int d = o & (D_MODEL - 1);
    int f = (o >> 9) & (D_FF - 1);
    int e = o >> 20;
    w1t[o] = f2bf(w1[((size_t)e * D_MODEL + d) * D_FF + f]);
  }
  {
    int f = o & (D_FF - 1);
    int d = (o >> 11) & (D_MODEL - 1);
    int e = o >> 20;
    w2t[o] = f2bf(w2[((size_t)e * D_FF + f) * D_MODEL + d]);
  }
}

// ---------------------------------------------------------------------------
// Kernel 4: fused per-expert FFN. grid = (128 token tiles, 8 experts),
// block = 256 = 8 wave32.  TDM double-buffered weight staging, bf16 WMMA,
// f32 accumulation, atomic expert combine.
// ---------------------------------------------------------------------------
__global__ __launch_bounds__(256, 1)
void moe_ffn_kernel(const unsigned short* __restrict__ xb,
                    const unsigned short* __restrict__ w1t,
                    const float* __restrict__ b1,
                    const unsigned short* __restrict__ w2t,
                    const float* __restrict__ b2,
                    const float* __restrict__ rw,
                    float* __restrict__ out) {
  __shared__ unsigned short sX[M_TILE * SX_STRIDE];       // x tile
  __shared__ unsigned short sH[M_TILE * SH_STRIDE];       // gelu(h) chunk
  __shared__ unsigned short sW[2][D_MODEL * SW_STRIDE];   // weight double buffer

  const int tid  = threadIdx.x;
  const int wave = tid >> 5;
  const int lane = tid & 31;
  const int hi   = (lane >> 4) & 1;
  const int ln   = lane & 15;
  const int e    = blockIdx.y;
  const int tokenBase = blockIdx.x * M_TILE;

  const unsigned short* w1e = w1t + (size_t)e * D_FF * D_MODEL;    // [f][d]
  const unsigned short* w2e = w2t + (size_t)e * D_MODEL * D_FF;    // [d][f]

  // --- Stage x tile via TDM: rows 1024B + 16B pad -> stride 520 elems ---
  if (wave == 0) {
    tdm_load_2d((unsigned)(unsigned long long)&sX[0],
                xb + (size_t)tokenBase * D_MODEL,
                /*tensor_d0=*/D_MODEL, /*tensor_d1=*/N_TOK,
                /*tile_d0=*/D_MODEL, /*tile_d1=*/M_TILE,
                /*stride0=*/D_MODEL,
                /*pad_icode=*/7 /*256 DW*/, /*pad_dw=*/4 /*16B*/);
    __builtin_amdgcn_s_wait_tensorcnt(0);
  }
  __syncthreads();

  const int rowBlk = (wave >> 1) * 16;
  const int half   = wave & 1;
  v8f oacc[16];
#pragma unroll
  for (int t = 0; t < 16; ++t) oacc[t] = vzero8();

  for (int fc = 0; fc < D_FF / F_TILE; ++fc) {
    const int f0 = fc * F_TILE;

    // ------------- Phase 1: H = gelu(X @ W1 + b1), 64 x 128 -------------
    v8f hacc[4];
#pragma unroll
    for (int t = 0; t < 4; ++t) hacc[t] = vzero8();

    // pre-issue chunk 0 (w1t rows f0.., cols ks*32.. ; 64B rows + 16B pad)
    __syncthreads();
    if (wave == 0)
      tdm_load_2d((unsigned)(unsigned long long)&sW[0][0],
                  w1e + (size_t)f0 * D_MODEL,
                  D_MODEL, D_FF, KSTEP, F_TILE, D_MODEL, 3, 4);

    for (int ks = 0; ks < D_MODEL / KSTEP; ++ks) {
      if (wave == 0) {
        if (ks + 1 < D_MODEL / KSTEP) {
          tdm_load_2d((unsigned)(unsigned long long)&sW[(ks + 1) & 1][0],
                      w1e + (size_t)f0 * D_MODEL + (ks + 1) * KSTEP,
                      D_MODEL, D_FF, KSTEP, F_TILE, D_MODEL, 3, 4);
          __builtin_amdgcn_s_wait_tensorcnt(1);
        } else {
          __builtin_amdgcn_s_wait_tensorcnt(0);
        }
      }
      __syncthreads();  // buf[ks&1] ready for everyone

      const unsigned short* wb = &sW[ks & 1][0];
      v16bf a = load_frag(&sX[(rowBlk + ln) * SX_STRIDE + ks * KSTEP + hi * 8]);
#pragma unroll
      for (int t = 0; t < 4; ++t) {
        v16bf b = load_frag(&wb[(half * 64 + t * 16 + ln) * SW_STRIDE + hi * 8]);
        hacc[t] = __builtin_amdgcn_wmma_f32_16x16x32_bf16(
            false, a, false, b, (short)0, hacc[t], false, false);
      }
      __syncthreads();  // readers done before buf is overwritten again
    }

    // bias + exact gelu, write H chunk to LDS as bf16
#pragma unroll
    for (int t = 0; t < 4; ++t) {
      int col = half * 64 + t * 16 + ln;
      float bias = b1[e * D_FF + f0 + col];
#pragma unroll
      for (int v = 0; v < 8; ++v) {
        int m = rowBlk + hi * 8 + v;
        float h = hacc[t][v] + bias;
        h = 0.5f * h * (1.0f + erff(h * 0.70710678118f));
        sH[m * SH_STRIDE + col] = f2bf(h);
      }
    }

    // ------------- Phase 2: OUT += H @ W2, 64 x 512 -------------
    __syncthreads();  // sH visible; previous sW readers done
    if (wave == 0)
      tdm_load_2d((unsigned)(unsigned long long)&sW[0][0],
                  w2e + (size_t)f0,
                  D_FF, D_MODEL, KSTEP, D_MODEL, D_FF, 3, 4);

    for (int ks2 = 0; ks2 < F_TILE / KSTEP; ++ks2) {
      if (wave == 0) {
        if (ks2 + 1 < F_TILE / KSTEP) {
          tdm_load_2d((unsigned)(unsigned long long)&sW[(ks2 + 1) & 1][0],
                      w2e + (size_t)f0 + (ks2 + 1) * KSTEP,
                      D_FF, D_MODEL, KSTEP, D_MODEL, D_FF, 3, 4);
          __builtin_amdgcn_s_wait_tensorcnt(1);
        } else {
          __builtin_amdgcn_s_wait_tensorcnt(0);
        }
      }
      __syncthreads();

      const unsigned short* wb = &sW[ks2 & 1][0];
      v16bf a = load_frag(&sH[(rowBlk + ln) * SH_STRIDE + ks2 * KSTEP + hi * 8]);
#pragma unroll
      for (int t = 0; t < 16; ++t) {
        v16bf b = load_frag(&wb[(half * 256 + t * 16 + ln) * SW_STRIDE + hi * 8]);
        oacc[t] = __builtin_amdgcn_wmma_f32_16x16x32_bf16(
            false, a, false, b, (short)0, oacc[t], false, false);
      }
      __syncthreads();
    }
  }

  // ------------- Epilogue: + b2, * route weight, atomic combine -------------
#pragma unroll
  for (int t = 0; t < 16; ++t) {
    int n = half * 256 + t * 16 + ln;
    float bias = b2[e * D_MODEL + n];
#pragma unroll
    for (int v = 0; v < 8; ++v) {
      int m = rowBlk + hi * 8 + v;
      int token = tokenBase + m;
      float val = (oacc[t][v] + bias) * rw[token * N_EXP + e];
      unsafeAtomicAdd(&out[(size_t)token * D_MODEL + n], val);
    }
  }
}

// ---------------------------------------------------------------------------
extern "C" void kernel_launch(void* const* d_in, const int* in_sizes, int n_in,
                              void* d_out, int out_size, void* d_ws, size_t ws_size,
                              hipStream_t stream) {
  const float* x  = (const float*)d_in[0];
  const float* w1 = (const float*)d_in[1];
  const float* b1 = (const float*)d_in[2];
  const float* w2 = (const float*)d_in[3];
  const float* b2 = (const float*)d_in[4];
  const float* wr = (const float*)d_in[5];
  const float* br = (const float*)d_in[6];

  float* out = (float*)d_out;                      // [8192 x 512]
  float* rw  = out + (size_t)N_TOK * D_MODEL;      // [8192 x 8]

  // workspace: bf16 x, bf16 transposed weights (~40 MB)
  unsigned short* xb  = (unsigned short*)d_ws;                 // 8192*512
  unsigned short* w1t = xb + (size_t)N_TOK * D_MODEL;          // 8*2048*512
  unsigned short* w2t = w1t + (size_t)N_EXP * D_FF * D_MODEL;  // 8*512*2048

  int n4 = (N_TOK * D_MODEL) / 4;
  moe_zero_kernel<<<(n4 + 255) / 256, 256, 0, stream>>>((float4*)out, n4);
  moe_router_kernel<<<(N_TOK + 255) / 256, 256, 0, stream>>>(x, wr, br, rw);

  int npair = (N_TOK * D_MODEL) / 2;
  moe_cvt_x_kernel<<<(npair + 255) / 256, 256, 0, stream>>>(x, (unsigned int*)xb, npair);
  int nw = N_EXP * D_FF * D_MODEL;
  moe_cvt_w_kernel<<<(nw + 255) / 256, 256, 0, stream>>>(w1, w2, w1t, w2t, nw);

  dim3 grid(N_TOK / M_TILE, N_EXP);
  moe_ffn_kernel<<<grid, 256, 0, stream>>>(xb, w1t, b1, w2t, b2, rw, out);
}
